// MambaG2G_2113123910055
// MI455X (gfx1250) — matrix operand order repaired and checked by reading, not compiled, not run on hardware
//
#include <hip/hip_runtime.h>
#include <hip/hip_bf16.h>
#include <math.h>

#define D_MODEL 64
#define D_STATE 16
#define D_CONV  4
#define D_INNER 128
#define DT_RANK 4
#define DIM_IN  5881
#define LIN_DIM 128
#define DIM_OUT 256
#define BATCH   64
#define SEQ     256
#define BT      (BATCH * SEQ)   // 16384

typedef __attribute__((ext_vector_type(16))) __bf16    bf16x16;
typedef __attribute__((ext_vector_type(8)))  float     f32x8;
typedef __attribute__((ext_vector_type(4)))  unsigned  u32x4;

union FragA { bf16x16 v; unsigned u[8]; };
union FragB { bf16x16 v; u32x4 q[2]; unsigned u[8]; };

// Fast pack: one v_perm_b32 concatenates the two high halfwords (truncating
// f32->bf16). Used in the hot GEMM loop.
static __device__ inline unsigned pack_bf16_fast(float a, float b) {
#if __has_builtin(__builtin_amdgcn_perm)
  // dst[15:0] = a[31:16], dst[31:16] = b[31:16]
  return __builtin_amdgcn_perm(__float_as_uint(b), __float_as_uint(a), 0x07060302u);
#else
  return (__float_as_uint(a) >> 16) | (__float_as_uint(b) & 0xFFFF0000u);
#endif
}

// Round-to-nearest-even pack; used only in the one-time weight prepack.
static __device__ inline unsigned pack_bf16_rne(float a, float b) {
  unsigned ua = __float_as_uint(a);
  unsigned ub = __float_as_uint(b);
  ua += 0x7FFFu + ((ua >> 16) & 1u);
  ub += 0x7FFFu + ((ub >> 16) & 1u);
  return (ua >> 16) | (ub & 0xFFFF0000u);
}

// Branchless guarded load: clamp index (always a valid address), select 0 via cndmask.
static __device__ inline float ldg_clamped(const float* __restrict__ row, int k, int K) {
  float v = row[(k < K) ? k : (K - 1)];
  return (k < K) ? v : 0.f;
}

// ---------------------------------------------------------------------------
// One-time weight prepack: B[N,K] fp32 row-major -> bf16 WMMA B-fragments.
// Fragment layout per (n-tile of 16, k-chunk of 32): 32 lanes x 8 dwords,
// lane = half*16 + r holds column N = nt*16 + r, K = kc*32 + half*16 + {0..15}
// packed as bf16 pairs (even K low, odd K high). K tail zero-padded.
// Storage: Bpk[((nt*nk + kc)*32 + lane)*8 + j]  (32B aligned per lane).
// ---------------------------------------------------------------------------
__global__ __launch_bounds__(256) void pack_b_kernel(
    const float* __restrict__ B, unsigned* __restrict__ Bpk,
    int N, int K, int ldb)
{
  const int nk = (K + 31) >> 5;
  const int total = (N >> 4) * nk * 32;
  const int idx = blockIdx.x * blockDim.x + threadIdx.x;
  if (idx >= total) return;
  const int lane = idx & 31;
  const int kc   = (idx >> 5) % nk;
  const int nt   = (idx >> 5) / nk;
  const int r    = lane & 15;
  const int half = lane >> 4;

  const float* Brow = B + (size_t)(nt * 16 + r) * ldb;
  const int kb0 = kc * 32 + (half << 4);

  FragB o;
  #pragma unroll
  for (int j = 0; j < 8; ++j) {
    const int kk = kb0 + 2 * j;
    const float b0 = ldg_clamped(Brow, kk,     K);
    const float b1 = ldg_clamped(Brow, kk + 1, K);
    o.u[j] = pack_bf16_rne(b0, b1);
  }
  u32x4* dst = (u32x4*)(Bpk + (size_t)idx * 8);
  dst[0] = o.q[0];
  dst[1] = o.q[1];
}

// Load one A fragment (16x32 MxK bf16) for row `Arow`, chunk base g0 (unguarded).
static __device__ inline FragA load_a_frag(const float* __restrict__ Arow, int g0) {
  FragA af;
  #pragma unroll
  for (int j = 0; j < 4; ++j) {
    af.u[j]     = pack_bf16_fast(Arow[g0 + 2*j],      Arow[g0 + 2*j + 1]);
    af.u[j + 4] = pack_bf16_fast(Arow[g0 + 8 + 2*j], Arow[g0 + 9 + 2*j]);
  }
  return af;
}

static __device__ inline FragA load_a_frag_tail(const float* __restrict__ Arow, int g0, int K) {
  FragA af;
  #pragma unroll
  for (int j = 0; j < 4; ++j) {
    af.u[j]     = pack_bf16_fast(ldg_clamped(Arow, g0 + 2*j,      K),
                                 ldg_clamped(Arow, g0 + 2*j + 1,  K));
    af.u[j + 4] = pack_bf16_fast(ldg_clamped(Arow, g0 + 8 + 2*j, K),
                                 ldg_clamped(Arow, g0 + 9 + 2*j, K));
  }
  return af;
}

// ---------------------------------------------------------------------------
// C[M,N] = scale * (A[M,K] x B^T) + bias[n], with B prepacked by pack_b_kernel.
// A row-major [M, lda] fp32, converted to bf16 in-registers (1 v_perm/pair).
// Each wave computes a 32x64 tile: 2 M-subtiles x 4 N-subtiles = 8 WMMAs per
// k-chunk sharing 4 B fragments. M%32==0, N%64==0.
// Unguarded main K loop + branchless tail.
// ---------------------------------------------------------------------------
__global__ __launch_bounds__(256) void gemm_nt_bf16_wmma(
    const float* __restrict__ A, const unsigned* __restrict__ Bpk,
    const float* __restrict__ bias, float* __restrict__ C,
    int M, int N, int K, int lda, int ldc, float scale)
{
  const int lane = threadIdx.x & 31;
  const int wave = blockIdx.x * (blockDim.x >> 5) + (threadIdx.x >> 5);
  const int mtiles = M >> 5;          // 32-row wave tiles
  const int ntiles = N >> 6;
  if (wave >= mtiles * ntiles) return;
  const int mt = wave / ntiles;
  const int nt = wave - mt * ntiles;
  const int m0 = mt << 5;
  const int n0 = nt << 6;
  const int r    = lane & 15;
  const int half = lane >> 4;

  const int nk     = (K + 31) >> 5;  // chunks stored in Bpk (incl. zero-padded tail)
  const int nkfull = K >> 5;         // unguarded chunks

  const f32x8 zero = {0.f,0.f,0.f,0.f,0.f,0.f,0.f,0.f};
  f32x8 acc[2][4];
  #pragma unroll
  for (int i = 0; i < 2; ++i)
    #pragma unroll
    for (int t = 0; t < 4; ++t) acc[i][t] = zero;

  const float* Arow0 = A + (size_t)(m0 + r) * lda;
  const float* Arow1 = A + (size_t)(m0 + 16 + r) * lda;
  const int nt0 = nt << 2;  // first 16-wide n-tile index

  for (int kc = 0; kc < nkfull; ++kc) {
    const int g0 = (kc << 5) + (half << 4);
    __builtin_prefetch(Arow0 + g0 + 64, 0, 1);
    __builtin_prefetch(Arow1 + g0 + 64, 0, 1);
    const FragA af0 = load_a_frag(Arow0, g0);
    const FragA af1 = load_a_frag(Arow1, g0);
    #pragma unroll
    for (int t = 0; t < 4; ++t) {
      const u32x4* bp = (const u32x4*)(Bpk +
          ((((size_t)(nt0 + t) * nk + kc) << 5) + lane) * 8);
      FragB bf;
      bf.q[0] = bp[0];
      bf.q[1] = bp[1];
      acc[0][t] = __builtin_amdgcn_wmma_f32_16x16x32_bf16(
          false, af0.v, false, bf.v, (short)0, acc[0][t], false, false);
      acc[1][t] = __builtin_amdgcn_wmma_f32_16x16x32_bf16(
          false, af1.v, false, bf.v, (short)0, acc[1][t], false, false);
    }
  }

  if (K & 31) {  // tail chunk: branchless clamped A loads; B side is pre-padded
    const int kc = nkfull;
    const int g0 = (kc << 5) + (half << 4);
    const FragA af0 = load_a_frag_tail(Arow0, g0, K);
    const FragA af1 = load_a_frag_tail(Arow1, g0, K);
    #pragma unroll
    for (int t = 0; t < 4; ++t) {
      const u32x4* bp = (const u32x4*)(Bpk +
          ((((size_t)(nt0 + t) * nk + kc) << 5) + lane) * 8);
      FragB bf;
      bf.q[0] = bp[0];
      bf.q[1] = bp[1];
      acc[0][t] = __builtin_amdgcn_wmma_f32_16x16x32_bf16(
          false, af0.v, false, bf.v, (short)0, acc[0][t], false, false);
      acc[1][t] = __builtin_amdgcn_wmma_f32_16x16x32_bf16(
          false, af1.v, false, bf.v, (short)0, acc[1][t], false, false);
    }
  }

  // store: C/D f32 layout: VGPR rr, lanes 0-15 -> (mb+rr, n), lanes 16-31 -> (mb+8+rr, n)
  #pragma unroll
  for (int i = 0; i < 2; ++i) {
    const int mbase = m0 + i * 16 + (half << 3);
    #pragma unroll
    for (int t = 0; t < 4; ++t) {
      const int ncol = n0 + t*16 + r;
      const float bv = bias ? bias[ncol] : 0.f;
      #pragma unroll
      for (int rr = 0; rr < 8; ++rr) {
        C[(size_t)(mbase + rr) * ldc + ncol] = acc[i][t][rr] * scale + bv;
      }
    }
  }
}

// ---------------------------------------------------------------------------
// One workgroup per batch element, one thread per d_inner channel.
// Causal depthwise conv + silu, x_dbl projection (via LDS), dt softplus,
// selective scan over T (states in registers), gating; accumulates
// ysum[b,d] = (1/T) * sum_t y_gated[b,t,d]  (mean folded in; W_out applied later).
// ---------------------------------------------------------------------------
__global__ __launch_bounds__(D_INNER) void mamba_scan_kernel(
    const float* __restrict__ xz,      // [B*T, 2*D_INNER] (x | gate)
    const float* __restrict__ conv_w,  // [D_INNER, 4]
    const float* __restrict__ conv_b,  // [D_INNER]
    const float* __restrict__ W_x,     // [36, D_INNER]
    const float* __restrict__ W_dt,    // [D_INNER, 4]
    const float* __restrict__ b_dt,    // [D_INNER]
    const float* __restrict__ A_log,   // [D_INNER, 16]
    const float* __restrict__ D_skip,  // [D_INNER]
    float* __restrict__ ysum_out)      // [B, D_INNER]
{
  const int b = blockIdx.x;
  const int d = threadIdx.x;  // 0..127

  __shared__ float s_xc[D_INNER];
  __shared__ float s_dbl[DT_RANK + 2 * D_STATE];            // 36
  __shared__ float s_Wx[(DT_RANK + 2 * D_STATE) * D_INNER]; // 36*128

  for (int i = d; i < (DT_RANK + 2 * D_STATE) * D_INNER; i += D_INNER)
    s_Wx[i] = W_x[i];

  const float w0 = conv_w[d * 4 + 0], w1 = conv_w[d * 4 + 1];
  const float w2 = conv_w[d * 4 + 2], w3 = conv_w[d * 4 + 3];
  const float cb = conv_b[d];
  const float bdt = b_dt[d];
  const float wdt0 = W_dt[d * 4 + 0], wdt1 = W_dt[d * 4 + 1];
  const float wdt2 = W_dt[d * 4 + 2], wdt3 = W_dt[d * 4 + 3];
  const float Dk = D_skip[d];
  float An[D_STATE];
  #pragma unroll
  for (int n = 0; n < D_STATE; ++n) An[n] = -expf(A_log[d * D_STATE + n]);

  float h[D_STATE];
  #pragma unroll
  for (int n = 0; n < D_STATE; ++n) h[n] = 0.f;

  float xm1 = 0.f, xm2 = 0.f, xm3 = 0.f;  // x[t-1], x[t-2], x[t-3]
  float ysum = 0.f;

  const float* xzb = xz + (size_t)b * SEQ * (2 * D_INNER);
  __syncthreads();

  for (int t = 0; t < SEQ; ++t) {
    const float xt = xzb[t * (2 * D_INNER) + d];
    const float gt = xzb[t * (2 * D_INNER) + D_INNER + d];
    float xc = xm3 * w0 + xm2 * w1 + xm1 * w2 + xt * w3 + cb;
    xc = xc / (1.f + expf(-xc));  // silu
    xm3 = xm2; xm2 = xm1; xm1 = xt;

    s_xc[d] = xc;
    __syncthreads();

    if (d < DT_RANK + 2 * D_STATE) {
      const float* wr = s_Wx + d * D_INNER;
      float a = 0.f;
      #pragma unroll 8
      for (int j = 0; j < D_INNER; ++j) a += wr[j] * s_xc[j];
      s_dbl[d] = a;
    }
    __syncthreads();

    float dtv = s_dbl[0] * wdt0 + s_dbl[1] * wdt1 + s_dbl[2] * wdt2 + s_dbl[3] * wdt3 + bdt;
    dtv = (dtv > 20.f) ? dtv : log1pf(expf(dtv));  // softplus

    float y = 0.f;
    #pragma unroll
    for (int n = 0; n < D_STATE; ++n) {
      const float dA  = expf(dtv * An[n]);
      const float dBx = dtv * s_dbl[DT_RANK + n] * xc;
      h[n] = dA * h[n] + dBx;
      y += h[n] * s_dbl[DT_RANK + D_STATE + n];
    }
    y += Dk * xc;
    y *= gt / (1.f + expf(-gt));  // * silu(gate)
    ysum += y;
    // next-iter s_dbl write is separated from this iter's reads by next sync
  }

  ysum_out[b * D_INNER + d] = ysum * (1.f / (float)SEQ);
}

// ---------------------------------------------------------------------------
// Heads: x = elu(tanh(e @ W_fc^T + b_fc)); mu = x@W_mu^T+b_mu;
//        sigma = elu(x@W_sig^T+b_sig)+1+1e-14.  One block per batch element.
// ---------------------------------------------------------------------------
__global__ __launch_bounds__(256) void heads_kernel(
    const float* __restrict__ e,       // [B, 64]
    const float* __restrict__ W_fc, const float* __restrict__ b_fc,
    const float* __restrict__ W_mu, const float* __restrict__ b_mu,
    const float* __restrict__ W_sig, const float* __restrict__ b_sig,
    float* __restrict__ out)           // x @ 0, mu @ 8192, sigma @ 24576
{
  const int b = blockIdx.x;
  const int tid = threadIdx.x;
  __shared__ float s_e[D_MODEL];
  __shared__ float s_x[LIN_DIM];

  if (tid < D_MODEL) s_e[tid] = e[b * D_MODEL + tid];
  __syncthreads();

  if (tid < LIN_DIM) {
    const float* w = W_fc + tid * D_MODEL;
    float a = b_fc[tid];
    #pragma unroll 8
    for (int j = 0; j < D_MODEL; ++j) a += s_e[j] * w[j];
    const float th = tanhf(a);
    const float xv = (th > 0.f) ? th : (expf(th) - 1.f);  // elu
    s_x[tid] = xv;
    out[b * LIN_DIM + tid] = xv;
  }
  __syncthreads();

  {
    const float* wm = W_mu + tid * LIN_DIM;
    const float* ws = W_sig + tid * LIN_DIM;
    float am = b_mu[tid];
    float as = b_sig[tid];
    #pragma unroll 8
    for (int j = 0; j < LIN_DIM; ++j) {
      const float xv = s_x[j];
      am += xv * wm[j];
      as += xv * ws[j];
    }
    out[BATCH * LIN_DIM + b * DIM_OUT + tid] = am;
    const float el = (as > 0.f) ? as : (expf(as) - 1.f);
    out[BATCH * LIN_DIM + BATCH * DIM_OUT + b * DIM_OUT + tid] = el + 1.f + 1e-14f;
  }
}

extern "C" void kernel_launch(void* const* d_in, const int* in_sizes, int n_in,
                              void* d_out, int out_size, void* d_ws, size_t ws_size,
                              hipStream_t stream)
{
  const float* input  = (const float*)d_in[0];
  const float* W_enc  = (const float*)d_in[1];
  const float* b_enc  = (const float*)d_in[2];
  const float* W_in   = (const float*)d_in[3];
  const float* conv_w = (const float*)d_in[4];
  const float* conv_b = (const float*)d_in[5];
  const float* W_x    = (const float*)d_in[6];
  const float* W_dt   = (const float*)d_in[7];
  const float* b_dt   = (const float*)d_in[8];
  const float* A_log  = (const float*)d_in[9];
  const float* D_skip = (const float*)d_in[10];
  const float* W_out  = (const float*)d_in[11];
  const float* W_fc   = (const float*)d_in[12];
  const float* b_fc   = (const float*)d_in[13];
  const float* W_mu   = (const float*)d_in[14];
  const float* b_mu   = (const float*)d_in[15];
  const float* W_sig  = (const float*)d_in[16];
  const float* b_sig  = (const float*)d_in[17];
  float* out = (float*)d_out;

  // packed-chunk counts
  const int nk_enc = (DIM_IN + 31) / 32;   // 184
  const int nk_in  = (D_MODEL + 31) / 32;  // 2

  // workspace layout (dwords):
  // z[16384,64] | xz[16384,256] | ysum[64,128] | e[64,64] | Bpk_enc | Bpk_in | Bpk_out
  float* zbuf    = (float*)d_ws;
  float* xzbuf   = zbuf + (size_t)BT * D_MODEL;
  float* ysumbuf = xzbuf + (size_t)BT * (2 * D_INNER);
  float* ebuf    = ysumbuf + (size_t)BATCH * D_INNER;
  unsigned* bpk_enc = (unsigned*)(ebuf + (size_t)BATCH * D_MODEL);
  unsigned* bpk_in  = bpk_enc + (size_t)(D_MODEL   / 16) * nk_enc * 32 * 8;
  unsigned* bpk_out = bpk_in  + (size_t)(2*D_INNER / 16) * nk_in  * 32 * 8;

  auto pack_blocks = [](int N, int K) {
    int total = (N >> 4) * ((K + 31) >> 5) * 32;
    return (total + 255) / 256;
  };
  auto gemm_blocks = [](int M, int N) {
    int waves = (M >> 5) * (N >> 6);   // 32-row wave tiles
    return (waves + 7) / 8;            // 8 waves (256 threads) per block
  };

  // 0) prepack weights into bf16 WMMA B-fragment layout (zero-padded K tails)
  pack_b_kernel<<<pack_blocks(D_MODEL, DIM_IN), 256, 0, stream>>>(
      W_enc, bpk_enc, D_MODEL, DIM_IN, DIM_IN);
  pack_b_kernel<<<pack_blocks(2 * D_INNER, D_MODEL), 256, 0, stream>>>(
      W_in, bpk_in, 2 * D_INNER, D_MODEL, D_MODEL);
  pack_b_kernel<<<pack_blocks(D_MODEL, D_INNER), 256, 0, stream>>>(
      W_out, bpk_out, D_MODEL, D_INNER, D_INNER);

  // 1) encoder: z = input @ W_enc^T + b_enc   [16384 x 5881] x [5881 x 64]
  gemm_nt_bf16_wmma<<<gemm_blocks(BT, D_MODEL), 256, 0, stream>>>(
      input, bpk_enc, b_enc, zbuf, BT, D_MODEL, DIM_IN, DIM_IN, D_MODEL, 1.f);

  // 2) xz = z @ W_in^T   [16384 x 64] x [64 x 256]
  gemm_nt_bf16_wmma<<<gemm_blocks(BT, 2 * D_INNER), 256, 0, stream>>>(
      zbuf, bpk_in, nullptr, xzbuf, BT, 2 * D_INNER, D_MODEL, D_MODEL, 2 * D_INNER, 1.f);

  // 3) conv + selective scan + gate -> time-mean pre-projection ysum[64,128]
  mamba_scan_kernel<<<BATCH, D_INNER, 0, stream>>>(
      xzbuf, conv_w, conv_b, W_x, W_dt, b_dt, A_log, D_skip, ysumbuf);

  // 4) e = ysum @ W_out^T   [64 x 128] x [128 x 64]
  gemm_nt_bf16_wmma<<<gemm_blocks(BATCH, D_MODEL), 256, 0, stream>>>(
      ysumbuf, bpk_out, nullptr, ebuf, BATCH, D_MODEL, D_INNER, D_INNER, D_MODEL, 1.f);

  // 5) heads -> (x, mu, sigma) concatenated in d_out
  heads_kernel<<<BATCH, 256, 0, stream>>>(
      ebuf, W_fc, b_fc, W_mu, b_mu, W_sig, b_sig, out);
}
